// Multi_Scale_Attention4_17523466568049
// MI455X (gfx1250) — matrix operand order repaired and checked by reading, not compiled, not run on hardware
//
#include <hip/hip_runtime.h>
#include <hip/hip_bf16.h>
#include <cstdint>

typedef _Float16 half_t;
typedef __attribute__((ext_vector_type(16))) _Float16 v16h;
typedef __attribute__((ext_vector_type(8)))  _Float16 v8h;
typedef __attribute__((ext_vector_type(8)))  float    v8f;

#define CCH   64
#define WPIX  128
#define HPIX  128
#define BATCH 8
#define HW    (HPIX * WPIX)          // 16384
#define CHW   (CCH * HW)
#define C3    192
#define C3HW  (C3 * HW)
#define KTOT  1728                   // 192*9

// padded leading dims (halves) for conflict-free b128 LDS access
#define LDR   136                    // row-major 64x128 buffers (x1,x2,xfp,xrp)
#define LDT   72                     // transposed 128x64 buffers + score
#define LDC   200                    // conv staging rows (192 ch + pad)
#define CBUF  15000                  // conv staging buffer halves (66*LDC + async slack)

// ---------------- WMMA fragment helpers (gfx1250, wave32) ----------------
// A fragment 16x32 f16, row-major source: two contiguous 8-half runs.
// lane l: m=l&15, kb=(l>>4)*8; halves j<8 -> k=kb+j ; j>=8 -> k=16+kb+(j-8)
__device__ __forceinline__ v16h load_frag_a(const half_t* __restrict__ src, int ld, int lane) {
  const half_t* row = src + (size_t)(lane & 15) * ld + ((lane >> 4) << 3);
  v8h lo = *(const v8h*)row;          // k = kb .. kb+7
  v8h hi = *(const v8h*)(row + 16);   // k = 16+kb .. 16+kb+7
  v16h r;
#pragma unroll
  for (int j = 0; j < 8; ++j) { r[j] = lo[j]; r[j + 8] = hi[j]; }
  return r;
}

// B fragment 32x16 f16 from K-MAJOR source M[n][k] (ld = K-stride):
// lane l: n=l&15, kb=(l>>4)*8; halves j<8 -> k=kb+j ; j>=8 -> k=16+kb+(j-8)
__device__ __forceinline__ v16h load_frag_bt(const half_t* __restrict__ src, int ld, int lane) {
  const half_t* row = src + (size_t)(lane & 15) * ld + ((lane >> 4) << 3);
  v8h lo = *(const v8h*)row;
  v8h hi = *(const v8h*)(row + 16);
  v16h r;
#pragma unroll
  for (int j = 0; j < 8; ++j) { r[j] = lo[j]; r[j + 8] = hi[j]; }
  return r;
}

__device__ __forceinline__ v8f wmma_f16(v16h a, v16h b, v8f c) {
  return __builtin_amdgcn_wmma_f32_16x16x32_f16(false, a, false, b, (short)0, c, false, false);
}

// Async 16-byte global -> LDS copy (ASYNCcnt-tracked, no VGPR round trip).
__device__ __forceinline__ void async_copy_b128(uint32_t lds_byte_off, const void* gaddr) {
  asm volatile("global_load_async_to_lds_b128 %0, %1, off"
               :: "v"(lds_byte_off), "v"(gaddr) : "memory");
}

// ---------------- prep kernels ----------------
__global__ void prep_small_kernel(
    const float* __restrict__ Wx,  const float* __restrict__ bx,
    const float* __restrict__ Wx1, const float* __restrict__ bx1,
    const float* __restrict__ Wx2, const float* __restrict__ bx2,
    const float* __restrict__ Wxf, const float* __restrict__ bxf,
    const float* __restrict__ Wxr, const float* __restrict__ bxr,
    const float* __restrict__ b_msa, const float* __restrict__ gamma,
    const float* __restrict__ beta,  const float* __restrict__ mean,
    const float* __restrict__ var,
    half_t* __restrict__ Wc, float* __restrict__ biasv,
    float* __restrict__ scale, float* __restrict__ biasc,
    half_t* __restrict__ zerobuf)
{
  const int tid = threadIdx.x;
  for (int i = tid; i < 4096; i += 256) {
    Wc[i]            = (half_t)Wx[i];
    Wc[3 * 4096 + i] = (half_t)Wxf[i];
    Wc[4 * 4096 + i] = (half_t)Wxr[i];
  }
  for (int i = tid; i < 4096; i += 256) {   // folded: Wc1=Wx1*Wx, Wc2=Wx2*Wx
    int o = i >> 6, c = i & 63;
    float s1 = 0.f, s2 = 0.f;
    for (int k = 0; k < 64; ++k) {
      float wk = Wx[k * 64 + c];
      s1 += Wx1[o * 64 + k] * wk;
      s2 += Wx2[o * 64 + k] * wk;
    }
    Wc[4096 + i]     = (half_t)s1;
    Wc[2 * 4096 + i] = (half_t)s2;
  }
  for (int o = tid; o < 64; o += 256) {
    float s1 = bx1[o], s2 = bx2[o];
    for (int k = 0; k < 64; ++k) {
      s1 += Wx1[o * 64 + k] * bx[k];
      s2 += Wx2[o * 64 + k] * bx[k];
    }
    biasv[o] = bx[o]; biasv[64 + o] = s1; biasv[128 + o] = s2;
    biasv[192 + o] = bxf[o]; biasv[256 + o] = bxr[o];
  }
  for (int o = tid; o < C3; o += 256) {
    float inv = rsqrtf(var[o] + 1e-4f);
    float sc  = gamma[o] * inv;
    scale[o] = sc;
    biasc[o] = (b_msa[o] - mean[o]) * sc + beta[o];
  }
  for (int i = tid; i < 128; i += 256) zerobuf[i] = (half_t)0.f;  // async pad source
}

// W_msa (o,i,ky,kx) f32 -> Wm[o][ky][kx][i] f16 (K contiguous per tap)
__global__ void prep_wm_kernel(const float* __restrict__ W_msa, half_t* __restrict__ Wm) {
  int i = blockIdx.x * blockDim.x + threadIdx.x;
  if (i >= C3 * C3 * 9) return;
  int ic = i % C3;
  int r  = i / C3;
  int kk = r % 9;
  int o  = r / 9;
  Wm[i] = (half_t)W_msa[((size_t)o * C3 + ic) * 9 + kk];
}

// ---------------- fused attention-row kernel ----------------
// One workgroup per (b,h). 8 waves / 256 threads. concat written NHWC f16.
__global__ __launch_bounds__(256)
void attn_rows_kernel(const float* __restrict__ x, const float* __restrict__ xf,
                      const float* __restrict__ xr,
                      const half_t* __restrict__ Wc, const float* __restrict__ biasv,
                      half_t* __restrict__ cat)
{
  extern __shared__ half_t smem[];
  half_t* sInT = smem;                    // [128][LDT] transposed input row
  half_t* sXpT = sInT + 128 * LDT;        // [128][LDT] xp transposed (B of feature)
  half_t* sX1  = sXpT + 128 * LDT;        // [64][LDR]
  half_t* sX2  = sX1  + 64 * LDR;
  half_t* sXfp = sX2  + 64 * LDR;
  half_t* sXrp = sXfp + 64 * LDR;
  half_t* sS   = sXrp + 64 * LDR;         // [64][LDT] score buffer

  const int tid = threadIdx.x, wave = tid >> 5, lane = tid & 31;
  const int bh = blockIdx.x;
  const int h = bh & (HPIX - 1);
  const int b = bh >> 7;

  const size_t rowoff  = (size_t)b * CHW + (size_t)h * WPIX;
  const size_t pixbase = (size_t)(b * HPIX + h) * WPIX;   // pixel index of (b,h,0)

  // ---- stage x row transposed ----
  for (int idx = tid; idx < 8192; idx += 256) {
    int c = idx >> 7, w = idx & 127;                      // coalesced global read
    sInT[(size_t)w * LDT + c] = (half_t)x[rowoff + (size_t)c * HW + w];
  }
  __syncthreads();

  // ---- x_org -> concat channels 128..191 (NHWC, vector stores) ----
  for (int idx = tid; idx < 1024; idx += 256) {
    int w = idx >> 3, cb = (idx & 7) << 3;
    v8h v = *(const v8h*)(sInT + (size_t)w * LDT + cb);
    *(v8h*)(cat + (pixbase + w) * C3 + 128 + cb) = v;
  }

  // ---- projections xp/x1/x2 = Wc{0,1,2} @ x_row + bias : 96 WMMA tiles ----
  for (int t = wave; t < 96; t += 8) {
    int g  = t / 32;
    int mt = (t & 31) >> 3;
    int nt = t & 7;
    const half_t* Wg = Wc + (size_t)g * 4096;
    v8f acc = {};
#pragma unroll
    for (int ks = 0; ks < 2; ++ks) {
      v16h a  = load_frag_a (Wg   + mt * 16 * 64  + ks * 32, 64,  lane);
      v16h bf = load_frag_bt(sInT + nt * 16 * LDT + ks * 32, LDT, lane);
      acc = wmma_f16(a, bf, acc);
    }
    const float* bv = biasv + g * 64 + mt * 16;
    int n = lane & 15, mb = (lane >> 4) << 3;
    if (g == 0) {
      v8h pk;
#pragma unroll
      for (int r = 0; r < 8; ++r) pk[r] = (half_t)(acc[r] + bv[mb + r]);
      *(v8h*)(sXpT + (size_t)(nt * 16 + n) * LDT + mt * 16 + mb) = pk;  // transposed
    } else {
      half_t* dst = (g == 1) ? sX1 : sX2;
#pragma unroll
      for (int r = 0; r < 8; ++r)
        dst[(size_t)(mt * 16 + mb + r) * LDR + nt * 16 + n] = (half_t)(acc[r] + bv[mb + r]);
    }
  }
  __syncthreads();

  // ---- xfp = Wxf @ xf_row, xrp = Wxr @ xr_row (sInT reused for staging) ----
  for (int pass = 0; pass < 2; ++pass) {
    const float* src = pass ? xr : xf;
    half_t* dstm = pass ? sXrp : sXfp;
    const half_t* Wg = Wc + (size_t)(3 + pass) * 4096;
    const float* bv0 = biasv + (3 + pass) * 64;
    for (int idx = tid; idx < 8192; idx += 256) {
      int c = idx >> 7, w = idx & 127;
      sInT[(size_t)w * LDT + c] = (half_t)src[rowoff + (size_t)c * HW + w];
    }
    __syncthreads();
    for (int t = wave; t < 32; t += 8) {
      int mt = t >> 3, nt = t & 7;
      v8f acc = {};
#pragma unroll
      for (int ks = 0; ks < 2; ++ks) {
        v16h a  = load_frag_a (Wg   + mt * 16 * 64  + ks * 32, 64,  lane);
        v16h bf = load_frag_bt(sInT + nt * 16 * LDT + ks * 32, LDT, lane);
        acc = wmma_f16(a, bf, acc);
      }
      int n = lane & 15, mb = (lane >> 4) << 3;
#pragma unroll
      for (int r = 0; r < 8; ++r)
        dstm[(size_t)(mt * 16 + mb + r) * LDR + nt * 16 + n] = (half_t)(acc[r] + bv0[mb + r]);
    }
    __syncthreads();
  }

  // ---- score & feature for (x1, xfp) then (x2, xrp) ----
  for (int pass = 0; pass < 2; ++pass) {
    half_t* sXq = pass ? sX2 : sX1;
    half_t* sP  = pass ? sXrp : sXfp;

    // score = sXq (64x128) @ sP^T : 16 tiles, K=128 (sP row-major == K-major B).
    for (int t = wave; t < 16; t += 8) {
      int mt = t >> 2, nt = t & 3;
      v8f acc = {};
#pragma unroll
      for (int ks = 0; ks < 4; ++ks) {
        v16h a  = load_frag_a (sXq + mt * 16 * LDR + ks * 32, LDR, lane);
        v16h bf = load_frag_bt(sP  + nt * 16 * LDR + ks * 32, LDR, lane);
        acc = wmma_f16(a, bf, acc);
      }
      int n = lane & 15, mb = (lane >> 4) << 3;
#pragma unroll
      for (int r = 0; r < 8; ++r)
        sS[(size_t)(mt * 16 + mb + r) * LDT + nt * 16 + n] = (half_t)acc[r];
    }
    __syncthreads();

    // feature = score (64x64) @ xp (64x128): 32 tiles; B from sXpT (K-major).
    for (int t = wave; t < 32; t += 8) {
      int mt = t >> 3, nt = t & 7;
      v8f acc = {};
#pragma unroll
      for (int ks = 0; ks < 2; ++ks) {
        v16h a  = load_frag_a (sS   + mt * 16 * LDT + ks * 32, LDT, lane);
        v16h bf = load_frag_bt(sXpT + nt * 16 * LDT + ks * 32, LDT, lane);
        acc = wmma_f16(a, bf, acc);
      }
      int n = lane & 15, mb = (lane >> 4) << 3;
      v8h pk;
#pragma unroll
      for (int r = 0; r < 8; ++r) pk[r] = (half_t)acc[r];
      *(v8h*)(cat + (pixbase + nt * 16 + n) * C3 + pass * 64 + mt * 16 + mb) = pk;
    }
    __syncthreads();
  }
}

// ---------------- 3x3 conv 192->192 + BN + ReLU as implicit GEMM ----------------
// concat is NHWC f16. 64(M=o) x 64(N=w) tile per block, K=1728, 6144 blocks.
// Double-buffered async global->LDS staging (ASYNCcnt), overlapped with WMMA.
__global__ __launch_bounds__(256)
void conv3_kernel(const half_t* __restrict__ cat, const half_t* __restrict__ Wm,
                  const float* __restrict__ scale, const float* __restrict__ biasc,
                  const half_t* __restrict__ zerobuf, float* __restrict__ out)
{
  extern __shared__ half_t smem[];   // 2 x CBUF halves, each holds [66][LDC] K-major
  const int tid = threadIdx.x, wave = tid >> 5, lane = tid & 31;

  int blk = blockIdx.x;
  int mb = blk % 3;  int r1 = blk / 3;
  int wb = r1 & 1;   int r2 = r1 >> 1;
  int h  = r2 & (HPIX - 1);
  int b  = r2 >> 7;
  const int o0 = mb * 64, w0 = wb * 64;

  const int mt  = wave >> 1;          // both tiles of a wave share the A row
  const int nt0 = (wave & 1) * 2;

  const uint32_t lds0 = (uint32_t)(uintptr_t)smem;

  // issue exactly 7 async b128 copies per wave (uniform ASYNCcnt accounting);
  // invalid/pad chunks are sourced from the 0-filled page and land in slack rows.
  auto stage_async = [&](int buf, int hy) {
    bool hvalid = (hy >= 0) && (hy < HPIX);
    const half_t* srcrow = cat + (size_t)(b * HPIX + (hvalid ? hy : 0)) * WPIX * C3;
    uint32_t base = lds0 + (uint32_t)buf * (CBUF * 2);
#pragma unroll
    for (int it = 0; it < 7; ++it) {
      int idx = tid + it * 256;             // up to 1791; chunks beyond 1584 are pad
      int wl = idx / 24, ch = (idx % 24) << 3;
      int wg = w0 + wl - 1;
      bool valid = hvalid && (idx < 66 * 24) && (wg >= 0) && (wg < WPIX);
      const half_t* g = valid ? (srcrow + (size_t)wg * C3 + ch) : zerobuf;
      async_copy_b128(base + (uint32_t)(wl * LDC + ch) * 2u, g);
    }
  };

  v8f acc0 = {}, acc1 = {};
  const half_t* Arow = Wm + (size_t)(o0 + mt * 16) * KTOT;

  stage_async(0, h - 1);                       // prefetch row ky=0

  for (int ky = 0; ky < 3; ++ky) {
    int cur = ky & 1;
    if (ky < 2) {
      stage_async(1 - cur, h + ky);            // prefetch next row while computing
      asm volatile("s_wait_asynccnt 7" ::: "memory");   // current row complete
    } else {
      asm volatile("s_wait_asynccnt 0" ::: "memory");
    }
    __syncthreads();                           // all waves' staging visible

    const half_t* B_row = smem + (size_t)cur * CBUF;
    for (int kx = 0; kx < 3; ++kx) {
      const half_t* A_k = Arow + (ky * 3 + kx) * C3;
      const half_t* B_k = B_row + (size_t)kx * LDC;
#pragma unroll
      for (int kc = 0; kc < 6; ++kc) {
        v16h a  = load_frag_a (A_k + kc * 32, KTOT, lane);
        v16h b0 = load_frag_bt(B_k + (size_t)(nt0 * 16) * LDC + kc * 32, LDC, lane);
        v16h b1 = load_frag_bt(B_k + (size_t)((nt0 + 1) * 16) * LDC + kc * 32, LDC, lane);
        acc0 = wmma_f16(a, b0, acc0);
        acc1 = wmma_f16(a, b1, acc1);
      }
    }
    __syncthreads();                           // done reading cur before overwrite
  }

  // fused BN scale/shift + ReLU epilogue -> NCHW f32 output
  int n = lane & 15, mrow = (lane >> 4) << 3;
  float* obase = out + (size_t)b * C3HW + (size_t)h * WPIX + w0;
#pragma unroll
  for (int r = 0; r < 8; ++r) {
    int o = o0 + mt * 16 + mrow + r;
    float sc = scale[o], bb = biasc[o];
    float v0 = acc0[r] * sc + bb; v0 = v0 > 0.f ? v0 : 0.f;
    float v1 = acc1[r] * sc + bb; v1 = v1 > 0.f ? v1 : 0.f;
    obase[(size_t)o * HW + nt0 * 16 + n]       = v0;
    obase[(size_t)o * HW + (nt0 + 1) * 16 + n] = v1;
  }
}

// ---------------- launch ----------------
extern "C" void kernel_launch(void* const* d_in, const int* in_sizes, int n_in,
                              void* d_out, int out_size, void* d_ws, size_t ws_size,
                              hipStream_t stream) {
  const float* x     = (const float*)d_in[0];
  const float* xf    = (const float*)d_in[1];
  const float* xr    = (const float*)d_in[2];
  const float* Wx    = (const float*)d_in[3];
  const float* bx    = (const float*)d_in[4];
  const float* Wx1   = (const float*)d_in[5];
  const float* bx1   = (const float*)d_in[6];
  const float* Wx2   = (const float*)d_in[7];
  const float* bx2   = (const float*)d_in[8];
  const float* Wxf   = (const float*)d_in[9];
  const float* bxf   = (const float*)d_in[10];
  const float* Wxr   = (const float*)d_in[11];
  const float* bxr   = (const float*)d_in[12];
  const float* W_msa = (const float*)d_in[13];
  const float* b_msa = (const float*)d_in[14];
  const float* gam   = (const float*)d_in[15];
  const float* bet   = (const float*)d_in[16];
  const float* mean  = (const float*)d_in[17];
  const float* var   = (const float*)d_in[18];
  float* out = (float*)d_out;

  char* ws = (char*)d_ws;
  half_t* Wc    = (half_t*)ws;                       // 5*64*64 f16   = 40960 B
  float*  biasv = (float*)(ws + 40960);              // 5*64 f32
  float*  scale = (float*)(ws + 40960 + 1280);       // 192 f32
  float*  biasc = (float*)(ws + 40960 + 1280 + 768); // 192 f32
  half_t* Wm    = (half_t*)(ws + 43776);             // 192*9*192 f16 = 663552 B
  half_t* cat16 = (half_t*)(ws + 43776 + 663552);    // NHWC 8*128*128*192 f16
  half_t* zbuf  = (half_t*)(ws + 43776 + 663552 + (size_t)BATCH * HW * C3 * 2); // 256 B zeros

  prep_small_kernel<<<1, 256, 0, stream>>>(Wx, bx, Wx1, bx1, Wx2, bx2, Wxf, bxf,
                                           Wxr, bxr, b_msa, gam, bet, mean, var,
                                           Wc, biasv, scale, biasc, zbuf);
  prep_wm_kernel<<<(C3 * C3 * 9 + 255) / 256, 256, 0, stream>>>(W_msa, Wm);

  size_t attn_lds = (size_t)(2 * 128 * LDT + 4 * 64 * LDR + 64 * LDT) * sizeof(half_t);
  attn_rows_kernel<<<BATCH * HPIX, 256, attn_lds, stream>>>(x, xf, xr, Wc, biasv, cat16);

  size_t conv_lds = (size_t)(2 * CBUF) * sizeof(half_t);
  conv3_kernel<<<BATCH * HPIX * 2 * 3, 256, conv_lds, stream>>>(cat16, Wm, scale, biasc,
                                                                zbuf, out);
}
